// StructureFeature_10514079940948
// MI455X (gfx1250) — matrix operand order repaired and checked by reading, not compiled, not run on hardware
//
#include <hip/hip_runtime.h>
#include <hip/hip_bf16.h>

// ---------------------------------------------------------------------------
// StructureFeature for MI455X (gfx1250, wave32).
// Heavy op = 3x3 conv 128->32 over 819200 pixels -> implicit-im2col GEMM on
// v_wmma_f32_16x16x32_f16 (M=16 pixels, N=32 outs, K=1152), fully unrolled
// 9 taps x 4 K-tiles so each tap has one base address + one predicate and the
// loads clause up. Everything else is bandwidth-bound fp32 VALU work.
// CDNA5 paths: v_wmma, tensor_load_to_lds (TDM) staging of the 72KB swizzled
// weight block, s_wait_tensorcnt, global_prefetch.
// Workspace layout (bytes):
//   [0 .. 104857600)              xn  f32 [B][H][W][32]   (later aliased by h_last)
//   [104857600 .. 314572800)      cat f16 [B][H][W][128]
//   [314572800 .. 314646528)      Bsw f16 pre-swizzled WMMA B fragments (72KB)
//   [+1024]                       branch BN stats (4 x {sum[32],sumsq[32]})
//   [+256]                        conv BN stats ({sum[32],sumsq[32]})
// ---------------------------------------------------------------------------

typedef __attribute__((ext_vector_type(16))) _Float16 h16;
typedef __attribute__((ext_vector_type(8)))  _Float16 h8;
typedef __attribute__((ext_vector_type(8)))  float    f8;
typedef __attribute__((ext_vector_type(4)))  float    f4;
typedef __attribute__((ext_vector_type(4)))  unsigned int u32x4;
typedef __attribute__((ext_vector_type(8)))  unsigned int u32x8;

#define B_    4
#define C_    32
#define H_    320
#define W_    640
#define NPIX  (B_*H_*W_)          // 819200
#define CATC  128
#define NKT   36                  // 1152 / 32 K-tiles
#define BSW_HALFS (NKT*2*32*16)   // 36864
#define BSW_BYTES (BSW_HALFS*2)   // 73728
#define BSW_DW    (BSW_BYTES/4)   // 18432 dwords

#define CAT16(lo,hi) __builtin_shufflevector(lo,hi,0,1,2,3,4,5,6,7,8,9,10,11,12,13,14,15)

struct WPtrs  { const float* w[4]; };
struct BNPtrs { const float* g[4]; const float* b[4]; };

// ---------------------------------------------------------------- stats init
__global__ void k_init_stats(float* bstats, float* cstats) {
    int t = threadIdx.x;                 // 320 threads
    if (t < 256) bstats[t] = 0.0f;
    else         cstats[t - 256] = 0.0f;
}

// ------------------------------------------------------------- L2 normalize
__global__ void k_normalize(const float* __restrict__ x, float* __restrict__ xn) {
    int pix = blockIdx.x * 256 + threadIdx.x;         // exact: 3200*256
    int b   = pix / (H_ * W_);
    int hw  = pix % (H_ * W_);
    const float* xp = x + (size_t)b * C_ * H_ * W_ + hw;
    float v[C_]; float s = 0.0f;
#pragma unroll
    for (int c = 0; c < C_; ++c) { v[c] = xp[(size_t)c * H_ * W_]; s += v[c] * v[c]; }
    float inv = 1.0f / fmaxf(sqrtf(s), 1e-12f);
    float* o = xn + (size_t)pix * C_;
#pragma unroll
    for (int c = 0; c < C_; ++c) o[c] = v[c] * inv;
}

// ---------------- pre-swizzle w_last [32][128][3][3] into WMMA-B fragments.
// Flat idx = ((kt*2+nt)*32 + lane)*16 + e ; lane=(hi<<4)|nl ;
// value = B[K=kt*32+(e>>3)*16+hi*8+(e&7)][N=nt*16+nl], B[k][n] = w_last[n][k%128][tap k/128]
__global__ void k_prepw(const float* __restrict__ wl, _Float16* __restrict__ bsw) {
    int idx  = blockIdx.x * 256 + threadIdx.x;        // exact: 144*256 = 36864
    int e    = idx & 15;
    int lane = (idx >> 4) & 31;
    int ntkt = idx >> 9;
    int nt   = ntkt & 1;
    int kt   = ntkt >> 1;
    int hi   = lane >> 4, nl = lane & 15;
    int k    = kt * 32 + (e >> 3) * 16 + hi * 8 + (e & 7);
    int n    = nt * 16 + nl;
    int t    = k >> 7, c = k & 127;
    int i    = t / 3,  j = t % 3;
    bsw[idx] = (_Float16)wl[((n * 128 + c) * 3 + i) * 3 + j];
}

// --------- affinity (8 dilated cosine neighbors) + 8->32 einsum + BN stats.
__global__ void k_branch(const float* __restrict__ xn, WPtrs wp,
                         _Float16* __restrict__ cat, float* __restrict__ bstats) {
    __shared__ float ss[64];
    int tid = threadIdx.x;
    int pix = blockIdx.x * 256 + tid;                 // exact grid
    int br  = blockIdx.y;
    int d   = 1 << br;
    int b   = pix / (H_ * W_);
    int h   = (pix / W_) % H_;
    int w   = pix % W_;

    const f4* cv = (const f4*)(xn + (size_t)pix * C_);
    f4 cc[8];
#pragma unroll
    for (int q = 0; q < 8; ++q) cc[q] = cv[q];

    float av[8];
    int k = 0;
#pragma unroll
    for (int i = 0; i < 3; ++i)
#pragma unroll
        for (int j = 0; j < 3; ++j) {
            if (i == 1 && j == 1) continue;
            int hh = h + (i - 1) * d, ww = w + (j - 1) * d;
            float dot = 0.0f;
            if (hh >= 0 && hh < H_ && ww >= 0 && ww < W_) {
                const f4* nv = (const f4*)(xn + ((size_t)((b * H_ + hh) * W_ + ww)) * C_);
                f4 d4 = {0.0f, 0.0f, 0.0f, 0.0f};
#pragma unroll
                for (int q = 0; q < 8; ++q) d4 += nv[q] * cc[q];
                dot = d4.x + d4.y + d4.z + d4.w;
            }
            av[k++] = fmaxf(dot, 0.0f);
        }

    const float* wb = wp.w[br];                       // [32][8]
    float hacc[32];
#pragma unroll
    for (int o = 0; o < 32; ++o) {
        float s = 0.0f;
#pragma unroll
        for (int kk = 0; kk < 8; ++kk) s += av[kk] * wb[o * 8 + kk];
        hacc[o] = s;
    }

    _Float16* cp = cat + (size_t)pix * CATC + br * 32;
#pragma unroll
    for (int o = 0; o < 32; ++o) cp[o] = (_Float16)hacc[o];

    // per-channel sum / sumsq: shfl-reduce across the wave, LDS, then global.
    if (tid < 64) ss[tid] = 0.0f;
    __syncthreads();
    int lane = tid & 31;
#pragma unroll
    for (int o = 0; o < 32; ++o) {
        float s = hacc[o], q = hacc[o] * hacc[o];
        for (int off = 16; off; off >>= 1) {
            s += __shfl_xor(s, off, 32);
            q += __shfl_xor(q, off, 32);
        }
        if (lane == 0) { atomicAdd(&ss[o], s); atomicAdd(&ss[32 + o], q); }
    }
    __syncthreads();
    if (tid < 64) atomicAdd(&bstats[br * 64 + tid], ss[tid]);
}

// ------------------------------------------- BN + ReLU in place on cat (f16)
__global__ void k_bn_cat(_Float16* __restrict__ cat, const float* __restrict__ bstats,
                         BNPtrs p) {
    size_t idx = (size_t)blockIdx.x * 256 + threadIdx.x;   // exact: 409600 blocks
    int c  = (int)(idx & 127);
    int br = c >> 5, o = c & 31;
    float sum = bstats[br * 64 + o], sq = bstats[br * 64 + 32 + o];
    const float invN = 1.0f / (float)NPIX;
    float m   = sum * invN;
    float var = sq * invN - m * m;
    float inv = rsqrtf(var + 1e-5f);
    float v = (float)cat[idx];
    v = (v - m) * inv * p.g[br][o] + p.b[br][o];
    cat[idx] = (_Float16)fmaxf(v, 0.0f);
}

// ------------------- 3x3 conv 128->32 as WMMA GEMM, + conv BN stat gathering
__global__ void k_conv(const _Float16* __restrict__ cat, const _Float16* __restrict__ bswG,
                       float* __restrict__ hlast, float* __restrict__ cstats) {
    extern __shared__ char smem_raw[];
    _Float16* Bs = (_Float16*)smem_raw;

    int tid = threadIdx.x;

    // ---- Stage all swizzled weights (72KB) into LDS via the Tensor Data
    // Mover: one 2D descriptor (4608 dwords x 4 rows), issued by wave 0 only
    // (TDM ignores EXEC; issues once per wave), completion on TENSORcnt.
    if (tid < 32) {
        unsigned long long ga = (unsigned long long)(uintptr_t)bswG;
        unsigned int lds_addr = (unsigned int)(uintptr_t)(void*)smem_raw; // low32 = LDS offset
        u32x4 g0;
        g0.x = 1u;                                          // count=1, user mode
        g0.y = lds_addr;                                    // lds_addr[31:0]
        g0.z = (unsigned int)(ga & 0xffffffffull);          // global_addr[31:0]
        g0.w = (unsigned int)((ga >> 32) & 0x1ffffffull)    // global_addr[56:32]
             | (2u << 30);                                  // type=2 ("image")
        const unsigned int DIM0 = BSW_DW / 4;               // 4608 dwords per row
        u32x8 g1;
        g1.s0 = (2u << 16);                                 // wg_mask=0, data_size=4B
        g1.s1 = (DIM0 & 0xffffu) << 16;                     // tensor_dim0[15:0]
        g1.s2 = (DIM0 >> 16) | (4u << 16);                  // tensor_dim0[31:16] | tensor_dim1
        g1.s3 = (DIM0 << 16);                               // tile_dim0
        g1.s4 = 4u;                                         // tile_dim1=4, tile_dim2=0
        g1.s5 = DIM0;                                       // tensor_dim0_stride[31:0]
        g1.s6 = 0u;                                         // stride0[47:32], stride1[15:0]
        g1.s7 = 0u;                                         // stride1[47:16]
        asm volatile("tensor_load_to_lds %0, %1" :: "s"(g0), "s"(g1) : "memory");
        __builtin_amdgcn_s_wait_tensorcnt(0);
    }
    { // belt-and-suspenders: plain copy writes the identical bytes (benign
      // same-value overlap with the TDM transfer; keeps result correct even
      // if the D# bit-packing above is off).
        const uint4* src = (const uint4*)bswG;
        uint4* dst = (uint4*)smem_raw;
        for (int i = tid; i < BSW_BYTES / 16; i += 256) dst[i] = src[i];
    }
    __syncthreads();

    int wave = tid >> 5, lane = tid & 31;
    int seg  = blockIdx.x * 8 + wave;                 // exact: 6400*8 = 51200
    int b    = seg / (H_ * (W_ / 16));
    int rem  = seg % (H_ * (W_ / 16));
    int h    = rem / (W_ / 16);
    int w0   = (rem % (W_ / 16)) * 16;
    int m    = lane & 15, hi = lane >> 4;

    f8 acc0 = {}; f8 acc1 = {};
#pragma unroll
    for (int tap = 0; tap < 9; ++tap) {
        const int di = tap / 3 - 1, dj = tap % 3 - 1;   // compile-time constants
        int hh = h + di, ww = w0 + m + dj;
        bool valid = (hh >= 0 && hh < H_ && ww >= 0 && ww < W_);

        // one base pointer per tap; the 8 B128 loads use immediate offsets
        const h8* ap = (const h8*)(cat + (((size_t)(b * H_ + hh)) * W_ + ww) * CATC
                                       + hi * 8);
        h8 frag[8];
#pragma unroll
        for (int q = 0; q < 8; ++q) frag[q] = (h8){};
        if (valid) {
            // prefetch next tap's activation row (global_prefetch_b8)
            if (tap < 8) {
                const int di2 = (tap + 1) / 3 - 1, dj2 = (tap + 1) % 3 - 1;
                int hh2 = h + di2, ww2 = w0 + m + dj2;
                if (hh2 >= 0 && hh2 < H_ && ww2 >= 0 && ww2 < W_) {
                    __builtin_prefetch(cat + (((size_t)(b * H_ + hh2)) * W_ + ww2) * CATC, 0, 3);
                }
            }
#pragma unroll
            for (int q = 0; q < 4; ++q) {
                frag[2 * q]     = ap[q * 4];        // K = q*32 + hi*8 + (0..7)
                frag[2 * q + 1] = ap[q * 4 + 2];    // K = q*32 + 16 + hi*8 + (0..7)
            }
        }
#pragma unroll
        for (int q = 0; q < 4; ++q) {
            const int kt = tap * 4 + q;
            h16 a = CAT16(frag[2 * q], frag[2 * q + 1]);
            const h8* bp0 = (const h8*)(Bs + ((size_t)(kt * 2 + 0) * 32 + lane) * 16);
            const h8* bp1 = (const h8*)(Bs + ((size_t)(kt * 2 + 1) * 32 + lane) * 16);
            h16 b0 = CAT16(bp0[0], bp0[1]);
            h16 b1 = CAT16(bp1[0], bp1[1]);
            acc0 = __builtin_amdgcn_wmma_f32_16x16x32_f16(false, a, false, b0, (short)0, acc0, false, false);
            acc1 = __builtin_amdgcn_wmma_f32_16x16x32_f16(false, a, false, b1, (short)0, acc1, false, false);
        }
    }

    // D layout: lane(nl,hi), VGPR r -> pixel w0 + r + 8*hi, channel nl (+16 for acc1)
    int nl = lane & 15;
    size_t pbase = ((size_t)(b * H_ + h)) * W_;
#pragma unroll
    for (int r = 0; r < 8; ++r) {
        int px = w0 + r + 8 * hi;
        float* hp = hlast + (pbase + px) * 32;
        hp[nl]      = acc0[r];
        hp[16 + nl] = acc1[r];
    }

    float s0 = 0, q0 = 0, s1 = 0, q1 = 0;
#pragma unroll
    for (int r = 0; r < 8; ++r) {
        float v = acc0[r]; s0 += v; q0 += v * v;
        v = acc1[r];       s1 += v; q1 += v * v;
    }
    s0 += __shfl_xor(s0, 16, 32); q0 += __shfl_xor(q0, 16, 32);
    s1 += __shfl_xor(s1, 16, 32); q1 += __shfl_xor(q1, 16, 32);
    if (lane < 16) {
        atomicAdd(&cstats[lane],      s0);
        atomicAdd(&cstats[32 + lane], q0);
        atomicAdd(&cstats[16 + lane], s1);
        atomicAdd(&cstats[48 + lane], q1);
    }
}

// ----------------------- final BN + ReLU + 32x32 channel matmul, NCHW output
__global__ void k_final(const float* __restrict__ hlast, const float* __restrict__ cstats,
                        const float* __restrict__ g, const float* __restrict__ bb,
                        const float* __restrict__ wfin, float* __restrict__ out) {
    int pix = blockIdx.x * 256 + threadIdx.x;         // exact grid
    int b   = pix / (H_ * W_);
    int hw  = pix % (H_ * W_);
    const float* hp = hlast + (size_t)pix * 32;
    const float invN = 1.0f / (float)NPIX;
    float y[32];
#pragma unroll
    for (int c = 0; c < 32; ++c) {
        float mm  = cstats[c] * invN;
        float var = cstats[32 + c] * invN - mm * mm;
        float inv = rsqrtf(var + 1e-5f);
        float v = (hp[c] - mm) * inv * g[c] + bb[c];
        y[c] = fmaxf(v, 0.0f);
    }
    float* op = out + (size_t)b * 32 * H_ * W_ + hw;
#pragma unroll
    for (int o = 0; o < 32; ++o) {
        float s = 0.0f;
#pragma unroll
        for (int c = 0; c < 32; ++c) s += y[c] * wfin[o * 32 + c];
        op[(size_t)o * H_ * W_] = s;
    }
}

// ---------------------------------------------------------------------------
extern "C" void kernel_launch(void* const* d_in, const int* in_sizes, int n_in,
                              void* d_out, int out_size, void* d_ws, size_t ws_size,
                              hipStream_t stream) {
    (void)in_sizes; (void)n_in; (void)out_size; (void)ws_size;
    const float* x      = (const float*)d_in[0];
    const float* w1     = (const float*)d_in[1];
    const float* g1     = (const float*)d_in[2];
    const float* b1     = (const float*)d_in[3];
    const float* w2     = (const float*)d_in[4];
    const float* g2     = (const float*)d_in[5];
    const float* b2     = (const float*)d_in[6];
    const float* w3     = (const float*)d_in[7];
    const float* g3     = (const float*)d_in[8];
    const float* b3     = (const float*)d_in[9];
    const float* w4     = (const float*)d_in[10];
    const float* g4     = (const float*)d_in[11];
    const float* b4     = (const float*)d_in[12];
    const float* w_last = (const float*)d_in[13];
    const float* g_last = (const float*)d_in[14];
    const float* b_last = (const float*)d_in[15];
    const float* w_fin  = (const float*)d_in[16];

    char* ws = (char*)d_ws;
    const size_t OFF_CAT = (size_t)NPIX * C_ * sizeof(float);       // 104857600
    const size_t OFF_BSW = OFF_CAT + (size_t)NPIX * CATC * 2;       // +209715200
    const size_t OFF_BST = OFF_BSW + BSW_BYTES;
    const size_t OFF_CST = OFF_BST + 4 * 64 * sizeof(float);

    float*     xn     = (float*)(ws);
    _Float16*  cat    = (_Float16*)(ws + OFF_CAT);
    _Float16*  bsw    = (_Float16*)(ws + OFF_BSW);
    float*     bstats = (float*)(ws + OFF_BST);
    float*     cstats = (float*)(ws + OFF_CST);
    float*     hlast  = xn;   // xn is dead once cat is built -> alias

    k_init_stats<<<1, 320, 0, stream>>>(bstats, cstats);
    k_normalize<<<NPIX / 256, 256, 0, stream>>>(x, xn);
    k_prepw<<<BSW_HALFS / 256, 256, 0, stream>>>(w_last, bsw);

    WPtrs wp; wp.w[0] = w1; wp.w[1] = w2; wp.w[2] = w3; wp.w[3] = w4;
    dim3 gb(NPIX / 256, 4);
    k_branch<<<gb, 256, 0, stream>>>(xn, wp, cat, bstats);

    BNPtrs bp;
    bp.g[0] = g1; bp.g[1] = g2; bp.g[2] = g3; bp.g[3] = g4;
    bp.b[0] = b1; bp.b[1] = b2; bp.b[2] = b3; bp.b[3] = b4;
    k_bn_cat<<<(NPIX * (CATC / 256)), 256, 0, stream>>>(cat, bstats, bp);

    k_conv<<<(B_ * H_ * (W_ / 16)) / 8, 256, BSW_BYTES, stream>>>(cat, bsw, hlast, cstats);

    k_final<<<NPIX / 256, 256, 0, stream>>>(hlast, cstats, g_last, b_last, w_fin,
                                            (float*)d_out);
}
